// QuantizedLinear_45810121179303
// MI455X (gfx1250) — matrix operand order repaired and checked by reading, not compiled, not run on hardware
//
#include <hip/hip_runtime.h>

typedef __attribute__((ext_vector_type(16))) _Float16 v16h;
typedef __attribute__((ext_vector_type(8)))  float    v8f;

#define K_DIM 4096
#define N_DIM 11008
#define M_DIM 16

// ---------------------------------------------------------------------------
// Kernel 0: one-shot x fp32 -> f16 conversion into workspace (16x4096 = 128KB)
// Each thread converts 8 contiguous elements: 2x b128 load, 1x b128 store.
// ---------------------------------------------------------------------------
__global__ __launch_bounds__(256) void x_to_f16(
    const float* __restrict__ X, _Float16* __restrict__ Xh)
{
    const int i = (blockIdx.x * 256 + threadIdx.x) * 8;   // 8192 threads total
    const float4* src = reinterpret_cast<const float4*>(X + i);
    const float4 x0 = src[0], x1 = src[1];
    union { _Float16 h[8]; int4 q; } u;
    u.h[0] = (_Float16)x0.x; u.h[1] = (_Float16)x0.y;
    u.h[2] = (_Float16)x0.z; u.h[3] = (_Float16)x0.w;
    u.h[4] = (_Float16)x1.x; u.h[5] = (_Float16)x1.y;
    u.h[6] = (_Float16)x1.z; u.h[7] = (_Float16)x1.w;
    *reinterpret_cast<int4*>(Xh + i) = u.q;
}

// ---------------------------------------------------------------------------
// Main GEMM. Each block (8 waves, wave32) computes a 16x32 tile of Y.
//   wave>>2 : N-subtile (0..1), 16 outputs each
//   wave&3  : K-slice   (0..3), 1024 deep each
// A = x tile (f16, pre-converted), B = W^T tile (int8-in-int32 -> f16, exact),
// f32 accumulation via v_wmma_f32_16x16x32_f16, LDS cross-wave K reduction,
// fused per-channel scale + bias epilogue.
// ---------------------------------------------------------------------------
__global__ __launch_bounds__(256) void qlin_wmma_f16(
    const _Float16* __restrict__ Xh,  // [16, 4096] f16 (workspace)
    const int*      __restrict__ W,   // [11008, 4096] int8-in-int32
    const float*    __restrict__ S,   // [11008] per-channel scale
    const float*    __restrict__ B,   // [11008] bias
    float*          __restrict__ Y)   // [16, 11008] fp32
{
    const int lane   = threadIdx.x & 31;
    const int wave   = threadIdx.x >> 5;
    const int half   = lane >> 4;
    const int l16    = lane & 15;
    const int sub    = wave >> 2;
    const int kslice = wave & 3;

    const int n0  = blockIdx.x * 32 + sub * 16;
    const int col = n0 + l16;

    const int kbeg = kslice * (K_DIM / 4);
    const int kend = kbeg + (K_DIM / 4);

    const _Float16* xrow = Xh + (size_t)l16 * K_DIM;
    const int*      wrow = W  + (size_t)col * K_DIM;

    v8f c = {};

    for (int kk = kbeg; kk < kend; kk += 32) {
        // ---- A fragment (f16, no conversion): contiguous k order, k = half*16 + s (same as B) ----
        // a[0..7] = K at kk + half*16, a[8..15] = K at kk + half*16 + 8
        union { v16h v; struct { int4 lo; int4 hi; } q; } ua;
        ua.q.lo = *reinterpret_cast<const int4*>(xrow + kk + half * 16);
        ua.q.hi = *reinterpret_cast<const int4*>(xrow + kk + half * 16 + 8);

        // ---- B fragment: w[col][kk + half*16 .. +15] int32 -> f16 ----
        const int4* wp = reinterpret_cast<const int4*>(wrow + kk + half * 16);
        const int4 w0 = wp[0], w1 = wp[1], w2 = wp[2], w3 = wp[3];
        v16h b;
        b[0]  = (_Float16)w0.x; b[1]  = (_Float16)w0.y;
        b[2]  = (_Float16)w0.z; b[3]  = (_Float16)w0.w;
        b[4]  = (_Float16)w1.x; b[5]  = (_Float16)w1.y;
        b[6]  = (_Float16)w1.z; b[7]  = (_Float16)w1.w;
        b[8]  = (_Float16)w2.x; b[9]  = (_Float16)w2.y;
        b[10] = (_Float16)w2.z; b[11] = (_Float16)w2.w;
        b[12] = (_Float16)w3.x; b[13] = (_Float16)w3.y;
        b[14] = (_Float16)w3.z; b[15] = (_Float16)w3.w;

        c = __builtin_amdgcn_wmma_f32_16x16x32_f16(false, ua.v, false, b,
                                                   (short)0, c, false, false);
    }

    __shared__ float red[8 * 256];
    if (kslice != 0) {
        float* dst = red + (wave * 256 + lane * 8);
        #pragma unroll
        for (int r = 0; r < 8; ++r) dst[r] = c[r];
    }
    __syncthreads();

    if (kslice == 0) {
        #pragma unroll
        for (int p = 1; p < 4; ++p) {
            const float* src = red + ((wave + p) * 256 + lane * 8);
            #pragma unroll
            for (int r = 0; r < 8; ++r) c[r] += src[r];
        }
        const float sc = S[col];
        const float bi = B[col];
        float* op = Y + (size_t)(half * 8) * N_DIM + col;   // row M = half*8 + r
        #pragma unroll
        for (int r = 0; r < 8; ++r)
            op[(size_t)r * N_DIM] = c[r] * sc + bi;
    }
}

// ---------------------------------------------------------------------------
// Fallback (ws too small): identical GEMM but converts x fp32->f16 in-loop.
// ---------------------------------------------------------------------------
__global__ __launch_bounds__(256) void qlin_wmma_f16_nows(
    const float* __restrict__ X,
    const int*   __restrict__ W,
    const float* __restrict__ S,
    const float* __restrict__ B,
    float*       __restrict__ Y)
{
    const int lane   = threadIdx.x & 31;
    const int wave   = threadIdx.x >> 5;
    const int half   = lane >> 4;
    const int l16    = lane & 15;
    const int sub    = wave >> 2;
    const int kslice = wave & 3;

    const int n0  = blockIdx.x * 32 + sub * 16;
    const int col = n0 + l16;

    const int kbeg = kslice * (K_DIM / 4);
    const int kend = kbeg + (K_DIM / 4);

    const float* xrow = X + (size_t)l16 * K_DIM;
    const int*   wrow = W + (size_t)col * K_DIM;

    v8f c = {};

    for (int kk = kbeg; kk < kend; kk += 32) {
        const float4* xa0 = reinterpret_cast<const float4*>(xrow + kk + half * 16);
        const float4* xa1 = reinterpret_cast<const float4*>(xrow + kk + half * 16 + 8);
        const float4 x0 = xa0[0], x1 = xa0[1];
        const float4 x2 = xa1[0], x3 = xa1[1];
        v16h a;
        a[0]  = (_Float16)x0.x; a[1]  = (_Float16)x0.y;
        a[2]  = (_Float16)x0.z; a[3]  = (_Float16)x0.w;
        a[4]  = (_Float16)x1.x; a[5]  = (_Float16)x1.y;
        a[6]  = (_Float16)x1.z; a[7]  = (_Float16)x1.w;
        a[8]  = (_Float16)x2.x; a[9]  = (_Float16)x2.y;
        a[10] = (_Float16)x2.z; a[11] = (_Float16)x2.w;
        a[12] = (_Float16)x3.x; a[13] = (_Float16)x3.y;
        a[14] = (_Float16)x3.z; a[15] = (_Float16)x3.w;

        const int4* wp = reinterpret_cast<const int4*>(wrow + kk + half * 16);
        const int4 w0 = wp[0], w1 = wp[1], w2 = wp[2], w3 = wp[3];
        v16h b;
        b[0]  = (_Float16)w0.x; b[1]  = (_Float16)w0.y;
        b[2]  = (_Float16)w0.z; b[3]  = (_Float16)w0.w;
        b[4]  = (_Float16)w1.x; b[5]  = (_Float16)w1.y;
        b[6]  = (_Float16)w1.z; b[7]  = (_Float16)w1.w;
        b[8]  = (_Float16)w2.x; b[9]  = (_Float16)w2.y;
        b[10] = (_Float16)w2.z; b[11] = (_Float16)w2.w;
        b[12] = (_Float16)w3.x; b[13] = (_Float16)w3.y;
        b[14] = (_Float16)w3.z; b[15] = (_Float16)w3.w;

        c = __builtin_amdgcn_wmma_f32_16x16x32_f16(false, a, false, b,
                                                   (short)0, c, false, false);
    }

    __shared__ float red[8 * 256];
    if (kslice != 0) {
        float* dst = red + (wave * 256 + lane * 8);
        #pragma unroll
        for (int r = 0; r < 8; ++r) dst[r] = c[r];
    }
    __syncthreads();

    if (kslice == 0) {
        #pragma unroll
        for (int p = 1; p < 4; ++p) {
            const float* src = red + ((wave + p) * 256 + lane * 8);
            #pragma unroll
            for (int r = 0; r < 8; ++r) c[r] += src[r];
        }
        const float sc = S[col];
        const float bi = B[col];
        float* op = Y + (size_t)(half * 8) * N_DIM + col;
        #pragma unroll
        for (int r = 0; r < 8; ++r)
            op[(size_t)r * N_DIM] = c[r] * sc + bi;
    }
}

extern "C" void kernel_launch(void* const* d_in, const int* in_sizes, int n_in,
                              void* d_out, int out_size, void* d_ws, size_t ws_size,
                              hipStream_t stream) {
    const float* x  = (const float*)d_in[0];   // [16, 4096]
    const int*   w  = (const int*)  d_in[1];   // [11008, 4096]
    const float* s  = (const float*)d_in[2];   // [11008]
    const float* bi = (const float*)d_in[3];   // [11008]
    float*       y  = (float*)d_out;           // [16, 11008]

    const size_t xh_bytes = (size_t)M_DIM * K_DIM * sizeof(_Float16);  // 128 KB
    dim3 grid(N_DIM / 32);   // 344 blocks x 8 waves = 2752 waves
    dim3 block(256);

    if (ws_size >= xh_bytes) {
        _Float16* xh = (_Float16*)d_ws;
        // 16*4096 / (256 threads * 8 elems) = 32 blocks
        x_to_f16<<<dim3((M_DIM * K_DIM) / (256 * 8)), block, 0, stream>>>(x, xh);
        qlin_wmma_f16<<<grid, block, 0, stream>>>(xh, w, s, bi, y);
    } else {
        qlin_wmma_f16_nows<<<grid, block, 0, stream>>>(x, w, s, bi, y);
    }
}
